// clip_score_matrix_2070174237308
// MI455X (gfx1250) — matrix operand order repaired and checked by reading, not compiled, not run on hardware
//
#include <hip/hip_runtime.h>
#include <math.h>

#define NB 64
#define NL 512
#define ND 512

typedef _Float16 v16h __attribute__((ext_vector_type(16)));
typedef _Float16 v8h  __attribute__((ext_vector_type(8)));
typedef _Float16 v4h  __attribute__((ext_vector_type(4)));
typedef float    v8f  __attribute__((ext_vector_type(8)));
typedef unsigned int u32x4 __attribute__((ext_vector_type(4)));
typedef unsigned int u32x8 __attribute__((ext_vector_type(8)));

// ---------------- LDS map for kl_main_kernel (dynamic, 224 KB) ----------------
#define AX_OFF 0                    // A-tile xn: 32x512 f16  (32 KB)
#define AT_OFF 32768                // A-tile tn: 32x512 f16  (32 KB)
#define BB_OFF 65536                // B slice double buffer starts here
#define BROWB  80                   // 64B of K-slice data + 16B TDM pad per column row
#define BMATSZ (512 * BROWB)        // 40960 B per matrix per buffer
#define SMEM_TOTAL (BB_OFF + 4 * BMATSZ)   // 229376 B (224 KB)
#define SP_OFF 65536                // phase-2 alias: P tile 32x512 f32 (64 KB)
#define SQ_OFF 131072               // phase-2 alias: Q tile 32x512 f32 (64 KB)

// ---- A fragment (16x32 f16): lane<16 holds row (lane&15), K = {0..7, 16..23};
//      lane>=16 same row, K = {8..15, 24..31}. Caller pre-applies koffA.
static __device__ __forceinline__ v16h load_fragA(const _Float16* p) {
    v8h lo = *(const v8h*)p;
    v8h hi = *(const v8h*)(p + 16);
    return __builtin_shufflevector(lo, hi, 0,1,2,3,4,5,6,7,8,9,10,11,12,13,14,15);
}
// ---- B fragment (32x16 f16): lane<16 holds col (lane&15), K = {0..7, 16..23};
//      lane>=16 K = {8..15, 24..31}. Caller pre-applies koffB.
static __device__ __forceinline__ v16h load_fragB(const _Float16* p) {
    v8h lo = *(const v8h*)p;
    v8h hi = *(const v8h*)(p + 16);
    return __builtin_shufflevector(lo, hi, 0,1,2,3,4,5,6,7,8,9,10,11,12,13,14,15);
}

// ---- TDM: DMA a 512-row x 64B 2D tile (K-slice of one matrix) global -> LDS.
// D# per CDNA5 ISA. data_size=8B units. LDS rows padded
// 64B -> 80B (pad_interval=3: every 8x8B; pad_amount=3: 4 DWORDs) so that the
// per-column ds_load_b128 fragment reads are LDS-bank-conflict-free.
static __device__ __forceinline__ void tdm_load_slice(unsigned int lds_byte_off,
                                                      const _Float16* gptr) {
    unsigned long long ga = (unsigned long long)(uintptr_t)gptr;
    u32x4 g0;
    g0[0] = 1u;                                            // count=1 (valid), no gather
    g0[1] = lds_byte_off;                                  // lds_addr
    g0[2] = (unsigned int)ga;                              // global_addr[31:0]
    g0[3] = (unsigned int)((ga >> 32) & 0x1FFFFFFu) | (2u << 30); // addr[56:32] | type=2
    u32x8 g1;
    g1[0] = (3u << 16) | (1u << 20) | (3u << 22) | (3u << 25);
            // data_size=3 (8B) | pad_enable | pad_interval=3 (64B) | pad_amount=3 (16B)
    g1[1] = (8u << 16);                                    // tensor_dim0 = 8 units (64B row)
    g1[2] = (512u << 16);                                  // tensor_dim1 = 512 rows
    g1[3] = (8u << 16);                                    // tile_dim0 = 8 units
    g1[4] = 512u;                                          // tile_dim1 = 512, tile_dim2 = 0
    g1[5] = 128u;                                          // tensor_dim0_stride = 128 units (1024B)
    g1[6] = 0u;
    g1[7] = 0u;
    // 2-group form: VADDR2/VADDR3 = NULL (tensor up to 2D)
    asm volatile("tensor_load_to_lds %0, %1" :: "s"(g0), "s"(g1) : "memory");
}

// Kernel 0: per-batch valid count + zero the output accumulator.
__global__ __launch_bounds__(256)
void count_zero_kernel(const int* __restrict__ mask, int* __restrict__ counts,
                       float* __restrict__ out) {
    const int b   = blockIdx.x;
    const int tid = threadIdx.x;
    int c = (mask[b * NL + tid] > 0) + (mask[b * NL + tid + 256] > 0);
    #pragma unroll
    for (int o = 16; o > 0; o >>= 1) c += __shfl_xor(c, o, 32);
    __shared__ int red[8];
    if ((tid & 31) == 0) red[tid >> 5] = c;
    __syncthreads();
    if (tid == 0) {
        int tot = 0;
        #pragma unroll
        for (int i = 0; i < 8; ++i) tot += red[i];
        counts[b] = tot;
        if (b == 0) out[0] = 0.0f;
    }
}

// Kernel 1: l2-normalize one 512-float row, emit f16. blockIdx < NB*NL -> x, else target.
__global__ __launch_bounds__(128)
void normalize_kernel(const float* __restrict__ x, const float* __restrict__ t,
                      _Float16* __restrict__ xn, _Float16* __restrict__ tn) {
    const size_t row  = blockIdx.x;
    const int    tid  = threadIdx.x;
    const bool   isT  = row >= (size_t)NB * NL;
    const size_t r    = isT ? row - (size_t)NB * NL : row;
    const float* src  = (isT ? t : x) + r * ND;
    _Float16*    dst  = (isT ? tn : xn) + r * ND;

    float4 v = ((const float4*)src)[tid];
    float  s = v.x * v.x + v.y * v.y + v.z * v.z + v.w * v.w;
    #pragma unroll
    for (int o = 16; o > 0; o >>= 1) s += __shfl_xor(s, o, 32);
    __shared__ float red[4];
    __shared__ float invs;
    if ((tid & 31) == 0) red[tid >> 5] = s;
    __syncthreads();
    if (tid == 0) {
        float tot = red[0] + red[1] + red[2] + red[3];
        invs = 1.0f / fmaxf(sqrtf(tot), 1e-12f);
    }
    __syncthreads();
    const float inv = invs;
    v4h o;
    o[0] = (_Float16)(v.x * inv);
    o[1] = (_Float16)(v.y * inv);
    o[2] = (_Float16)(v.z * inv);
    o[3] = (_Float16)(v.w * inv);
    ((v4h*)dst)[tid] = o;
}

// Kernel 2: per (batch, 32-row tile): TDM-fed WMMA Gram rows for P and Q,
// then masked log_softmax/softmax/KL over 512 columns, reduce, atomicAdd.
// 2 waves/SIMD is the LDS-imposed occupancy ceiling -> tell the compiler.
__global__ __launch_bounds__(256, 2)
void kl_main_kernel(const _Float16* __restrict__ xn, const _Float16* __restrict__ tn,
                    const int* __restrict__ counts, float* __restrict__ out) {
    extern __shared__ char smem[];
    __shared__ float blockSum;

    const int tid  = threadIdx.x;
    const int lane = tid & 31;
    const int w    = tid >> 5;                 // wave id, 0..7
    const int b    = blockIdx.x >> 4;          // batch
    const int lt   = blockIdx.x & 15;          // 32-row tile index

    if (tid == 0) blockSum = 0.0f;

    const unsigned int smemBase = (unsigned int)(uintptr_t)smem;  // LDS byte offset
    const _Float16* gXbatch = xn + (size_t)b * NL * ND;
    const _Float16* gTbatch = tn + (size_t)b * NL * ND;

    // ---- Prologue: wave 0 DMAs the first B K-slice (both matrices) into buf0 ----
    if (w == 0) {
        tdm_load_slice(smemBase + BB_OFF + 0 * BMATSZ, gXbatch + 0);
        tdm_load_slice(smemBase + BB_OFF + 1 * BMATSZ, gTbatch + 0);
    }

    // ---- Stage the 32 A-side rows (contiguous 64 KB) of xn and tn into LDS ----
    {
        const uint4* srcX = (const uint4*)(gXbatch + (size_t)lt * 32 * ND);
        const uint4* srcT = (const uint4*)(gTbatch + (size_t)lt * 32 * ND);
        uint4* dX = (uint4*)(smem + AX_OFF);
        uint4* dT = (uint4*)(smem + AT_OFF);
        for (int i = tid; i < (32 * ND) / 8; i += 256) {
            dX[i] = srcX[i];
            dT[i] = srcT[i];
        }
    }
    if (w == 0) __builtin_amdgcn_s_wait_tensorcnt(0);
    __syncthreads();

    // ---- MMA phase: double-buffered TDM K-slices. Each wave: 4 column tiles x
    // 2 row tiles x 2 matrices = 16 WMMAs per k-step; every B fragment feeds
    // two WMMAs (row-tile reuse), cutting DS traffic per WMMA by 40%.
    const int koffA = (lane < 16) ? 0 : 8;
    const int rowA  = lane & 15;
    const int koffB = (lane < 16) ? 0 : 8;
    const int colB  = lane & 15;

    v8f accP[2][4] = {};
    v8f accQ[2][4] = {};

    int bColOff[4];
    #pragma unroll
    for (int j = 0; j < 4; ++j) {
        const int cj = (w * 4 + j) * 16 + colB;     // column this lane serves
        bColOff[j] = cj * BROWB + koffB * 2;        // byte offset inside a B matrix buffer
    }
    const _Float16* aXp0 = (const _Float16*)(smem + AX_OFF) + rowA * ND + koffA;
    const _Float16* aXp1 = aXp0 + 16 * ND;
    const _Float16* aTp0 = (const _Float16*)(smem + AT_OFF) + rowA * ND + koffA;
    const _Float16* aTp1 = aTp0 + 16 * ND;

    int cur = 0;
    for (int kstep = 0; kstep < 16; ++kstep) {
        const int k0 = kstep * 32;
        if (kstep < 15 && w == 0) {                 // DMA next slice into other buffer
            const unsigned int nb = BB_OFF + (1 - cur) * 2 * BMATSZ;
            tdm_load_slice(smemBase + nb + 0 * BMATSZ, gXbatch + (k0 + 32));
            tdm_load_slice(smemBase + nb + 1 * BMATSZ, gTbatch + (k0 + 32));
        }

        const char* bufX = smem + BB_OFF + cur * 2 * BMATSZ;
        const char* bufT = bufX + BMATSZ;

        v16h aX0 = load_fragA(aXp0 + k0);
        v16h aX1 = load_fragA(aXp1 + k0);
        v16h aT0 = load_fragA(aTp0 + k0);
        v16h aT1 = load_fragA(aTp1 + k0);
        #pragma unroll
        for (int j = 0; j < 4; ++j) {
            v16h bX = load_fragB((const _Float16*)(bufX + bColOff[j]));
            v16h bT = load_fragB((const _Float16*)(bufT + bColOff[j]));
            accP[0][j] = __builtin_amdgcn_wmma_f32_16x16x32_f16(
                false, aX0, false, bX, (short)0, accP[0][j], false, false);
            accP[1][j] = __builtin_amdgcn_wmma_f32_16x16x32_f16(
                false, aX1, false, bX, (short)0, accP[1][j], false, false);
            accQ[0][j] = __builtin_amdgcn_wmma_f32_16x16x32_f16(
                false, aT0, false, bT, (short)0, accQ[0][j], false, false);
            accQ[1][j] = __builtin_amdgcn_wmma_f32_16x16x32_f16(
                false, aT1, false, bT, (short)0, accQ[1][j], false, false);
        }

        if (kstep < 15) {
            if (w == 0) __builtin_amdgcn_s_wait_tensorcnt(0);
            __syncthreads();                        // next buffer ready; current fully read
            cur ^= 1;
        }
    }
    __syncthreads();   // all reads of B buffers done before aliasing as sP/sQ

    // ---- Spill C tiles: VGPR r holds M=r (lanes 0-15) / M=r+8 (lanes 16-31) ----
    float* sP = (float*)(smem + SP_OFF);
    float* sQ = (float*)(smem + SQ_OFF);
    const int mbase = (lane < 16) ? 0 : 8;
    #pragma unroll
    for (int rt = 0; rt < 2; ++rt) {
        #pragma unroll
        for (int j = 0; j < 4; ++j) {
            const int col = (w * 4 + j) * 16 + colB;
            #pragma unroll
            for (int r = 0; r < 8; ++r) {
                sP[(rt * 16 + mbase + r) * NL + col] = accP[rt][j][r];
                sQ[(rt * 16 + mbase + r) * NL + col] = accQ[rt][j][r];
            }
        }
    }
    __syncthreads();

    // ---- Softmax + KL phase: each wave handles 4 of the 32 rows ----
    const int m = counts[b];
    float wsum = 0.0f;
    for (int rr = 0; rr < 4; ++rr) {
        const int r    = w * 4 + rr;
        const int grow = lt * 32 + r;
        if (grow < m) {                             // row valid (implies m > 0)
            const float* Prow = sP + r * NL;
            const float* Qrow = sQ + r * NL;
            float mp = -1e30f, mq = -1e30f;
            for (int c = lane; c < m; c += 32) {
                mp = fmaxf(mp, Prow[c]);
                mq = fmaxf(mq, Qrow[c]);
            }
            #pragma unroll
            for (int o = 16; o > 0; o >>= 1) {
                mp = fmaxf(mp, __shfl_xor(mp, o, 32));
                mq = fmaxf(mq, __shfl_xor(mq, o, 32));
            }
            float sp = 0.0f, sq = 0.0f;
            for (int c = lane; c < m; c += 32) {
                sp += expf(Prow[c] - mp);
                sq += expf(Qrow[c] - mq);
            }
            #pragma unroll
            for (int o = 16; o > 0; o >>= 1) {
                sp += __shfl_xor(sp, o, 32);
                sq += __shfl_xor(sq, o, 32);
            }
            const float lzp = mp + logf(sp);
            const float lzq = mq + logf(sq);
            float kl = 0.0f;
            for (int c = lane; c < m; c += 32) {
                const float lq = Qrow[c] - lzq;
                const float lp = Prow[c] - lzp;
                kl += expf(lq) * (lq - lp);
            }
            #pragma unroll
            for (int o = 16; o > 0; o >>= 1) kl += __shfl_xor(kl, o, 32);
            wsum += kl;
        }
    }
    if (lane == 0) atomicAdd(&blockSum, wsum);
    __syncthreads();
    if (tid == 0) atomicAdd(out, blockSum * (1.0f / (float)NB));
}

extern "C" void kernel_launch(void* const* d_in, const int* in_sizes, int n_in,
                              void* d_out, int out_size, void* d_ws, size_t ws_size,
                              hipStream_t stream) {
    const float* x      = (const float*)d_in[0];
    const float* target = (const float*)d_in[1];
    const int*   mask   = (const int*)d_in[2];
    float*       out    = (float*)d_out;

    // Workspace: [counts: 64 ints, padded to 256B][xn: f16 B*L*D][tn: f16 B*L*D]
    char*      ws     = (char*)d_ws;
    int*       counts = (int*)ws;
    _Float16*  xn     = (_Float16*)(ws + 256);
    _Float16*  tn     = xn + (size_t)NB * NL * ND;

    count_zero_kernel<<<NB, 256, 0, stream>>>(mask, counts, out);
    normalize_kernel<<<2 * NB * NL, 128, 0, stream>>>(x, target, xn, tn);
    kl_main_kernel<<<NB * (NL / 32), 256, SMEM_TOTAL, stream>>>(xn, tn, counts, out);
}
